// GPT2Attention_2199023256294
// MI455X (gfx1250) — compile-verified, matrix operand order checked by reading, not run on hardware
//
#include <hip/hip_runtime.h>

// ---------------- problem constants ----------------
constexpr int kB  = 8;
constexpr int kS  = 1024;
constexpr int kD  = 1024;
constexpr int kH  = 16;
constexpr int kHD = 64;
constexpr float kScaling = 8.0f;          // 32/4
constexpr int kM    = kB * kS;            // 8192 rows
constexpr int kNqkv = 3 * kD;             // 3072

typedef __attribute__((ext_vector_type(8)))  float  v8f;
typedef __attribute__((ext_vector_type(8)))  __bf16 v8bf;
typedef __attribute__((ext_vector_type(16))) __bf16 v16bf;

// ---------------- helpers ----------------
static __device__ __forceinline__ __bf16 f2bf(float f) {
  unsigned u = __builtin_bit_cast(unsigned, f);
  u += 0x7fffu + ((u >> 16) & 1u);                 // round-to-nearest-even
  unsigned short h = (unsigned short)(u >> 16);
  return __builtin_bit_cast(__bf16, h);
}

static __device__ __forceinline__ v16bf cat16(v8bf lo, v8bf hi) {
  return __builtin_shufflevector(lo, hi, 0,1,2,3,4,5,6,7,8,9,10,11,12,13,14,15);
}

static __device__ __forceinline__ v8f wmma_bf16(v16bf a, v16bf b, v8f c) {
  return __builtin_amdgcn_wmma_f32_16x16x32_bf16(false, a, false, b, (short)0, c,
                                                 false, false);
}

// Load a K=32 fragment (A or B layout) from a 32-element-wide row.
// lanes 0-15 take K {0..7,16..23}; lanes 16-31 take K {8..15,24..31}.
static __device__ __forceinline__ v16bf load_row32(const __bf16* p, int half) {
  v8bf lo = *(const v8bf*)(p + half * 8);
  v8bf hi = *(const v8bf*)(p + 16 + half * 8);
  return cat16(lo, hi);
}

// Load two K=32 fragments from a contiguous 64-element row (q / k rows).
static __device__ __forceinline__ void load_row64(const __bf16* p, int half,
                                                  v16bf& f0, v16bf& f1) {
  v8bf c0 = *(const v8bf*)(p + 0);
  v8bf c1 = *(const v8bf*)(p + 8);
  v8bf c2 = *(const v8bf*)(p + 16);
  v8bf c3 = *(const v8bf*)(p + 24);
  v8bf c4 = *(const v8bf*)(p + 32);
  v8bf c5 = *(const v8bf*)(p + 40);
  v8bf c6 = *(const v8bf*)(p + 48);
  v8bf c7 = *(const v8bf*)(p + 56);
  f0 = half ? cat16(c1, c3) : cat16(c0, c2);
  f1 = half ? cat16(c5, c7) : cat16(c4, c6);
}

// ---------------- CDNA5 async global->LDS (ASYNCcnt path) ----------------
// ISA aperture rule: low 32 bits of a generic LDS pointer are the LDS byte addr.
static __device__ __forceinline__ unsigned lds_addr(const void* p) {
  return (unsigned)(unsigned long long)p;
}
static __device__ __forceinline__ void async_ld_b128(unsigned lds_off,
                                                     const __bf16* g) {
  asm volatile("global_load_async_to_lds_b128 %0, %1, off"
               :: "v"(lds_off), "v"(g) : "memory");
}
static __device__ __forceinline__ void wait_async0() {
  asm volatile("s_wait_asynccnt 0x0" ::: "memory");
}

// ---------------- conversion kernels ----------------
__global__ void cvt_kernel(const float* __restrict__ in,
                           __bf16* __restrict__ out, int n) {
  int i = blockIdx.x * 256 + threadIdx.x;
  if (i < n) out[i] = f2bf(in[i]);
}

// in: (K,N) row-major fp32 -> out: (N,K) row-major bf16 (transposed weights)
__global__ void cvtT_kernel(const float* __restrict__ in,
                            __bf16* __restrict__ out, int K, int N) {
  int i = blockIdx.x * 256 + threadIdx.x;
  if (i < N * K) {
    int nn = i / K, kk = i - nn * K;
    out[i] = f2bf(in[kk * N + nn]);
  }
}

// ---------------- LoRA tA = x . lora_A^T  (rank-8 output per row) ----------------
__global__ __launch_bounds__(256) void lora_ta_kernel(
    const float* __restrict__ x, const float* __restrict__ lora_A,
    float* __restrict__ tA) {
  __shared__ float red[256 * 8];
  const int m = blockIdx.x, tid = threadIdx.x;
  float p[8] = {};
  for (int d = tid; d < kD; d += 256) {
    float xv = x[m * kD + d];
#pragma unroll
    for (int j = 0; j < 8; ++j) p[j] += xv * lora_A[j * kD + d];
  }
#pragma unroll
  for (int j = 0; j < 8; ++j) red[tid * 8 + j] = p[j];
  __syncthreads();
  for (int s2 = 128; s2 > 0; s2 >>= 1) {
    if (tid < s2) {
#pragma unroll
      for (int j = 0; j < 8; ++j) red[tid * 8 + j] += red[(tid + s2) * 8 + j];
    }
    __syncthreads();
  }
  if (tid < 8) tA[m * 8 + tid] = red[tid];
}

// ---------------- shared GEMM tiling constants ----------------
constexpr int kLdsStride = 40;  // 32 + 8 pad: 16B aligned, avoids bank conflicts

// Stage one 128x32 A tile + 128x32 B tile into LDS via async DMA.
// Each thread issues 4 async b128 ops; no VGPR round-trip, tracked by ASYNCcnt.
static __device__ __forceinline__ void stage_tiles_async(
    __bf16* As, __bf16* Bs, const __bf16* __restrict__ gA,
    const __bf16* __restrict__ gB, int mbase, int nbase, int k0, int tid) {
  const int r0 = tid >> 2, part = tid & 3;  // r0 in [0,64), two row passes
  async_ld_b128(lds_addr(&As[r0 * kLdsStride + part * 8]),
                &gA[(mbase + r0) * kD + k0 + part * 8]);
  async_ld_b128(lds_addr(&As[(r0 + 64) * kLdsStride + part * 8]),
                &gA[(mbase + r0 + 64) * kD + k0 + part * 8]);
  async_ld_b128(lds_addr(&Bs[r0 * kLdsStride + part * 8]),
                &gB[(nbase + r0) * kD + k0 + part * 8]);
  async_ld_b128(lds_addr(&Bs[(r0 + 64) * kLdsStride + part * 8]),
                &gB[(nbase + r0 + 64) * kD + k0 + part * 8]);
}

// ---------------- fused QKV GEMM + bias + LoRA + head split ----------------
// C(8192x3072) = Xbf(8192x1024) @ Wbf^T ; 128x128 block tile, 32x64 wave tile,
// double-buffered async LDS staging.
__global__ __launch_bounds__(256) void qkv_gemm_kernel(
    const __bf16* __restrict__ xbf, const __bf16* __restrict__ wT,
    const float* __restrict__ b_qkv, const float* __restrict__ lora_B,
    const float* __restrict__ tA, __bf16* __restrict__ qbf,
    __bf16* __restrict__ kbf, __bf16* __restrict__ vtbf,
    float* __restrict__ kout, float* __restrict__ vout) {
  __shared__ __bf16 As[2][128 * kLdsStride];
  __shared__ __bf16 Bs[2][128 * kLdsStride];
  const int tid = threadIdx.x;
  const int lane = tid & 31, w = tid >> 5;
  const int half = lane >> 4, l16 = lane & 15;
  const int nblkN = kNqkv / 128;
  const int mbase = (blockIdx.x / nblkN) * 128;
  const int nbase = (blockIdx.x % nblkN) * 128;
  const int wm = (w & 3) * 32;   // wave rows [wm, wm+32)
  const int wn = (w >> 2) * 64;  // wave cols [wn, wn+64)
  v8f acc[2][4] = {};

  constexpr int NK = kD / 32;
  stage_tiles_async(As[0], Bs[0], xbf, wT, mbase, nbase, 0, tid);
  for (int kt = 0; kt < NK; ++kt) {
    const int cur = kt & 1;
    wait_async0();        // this wave's tile-kt DMA has landed in LDS
    __syncthreads();      // all waves' tile-kt landed; prev reads of buf^1 done
    if (kt + 1 < NK)
      stage_tiles_async(As[cur ^ 1], Bs[cur ^ 1], xbf, wT, mbase, nbase,
                        (kt + 1) * 32, tid);
    v16bf a0 = load_row32(&As[cur][(wm + l16) * kLdsStride], half);
    v16bf a1 = load_row32(&As[cur][(wm + 16 + l16) * kLdsStride], half);
#pragma unroll
    for (int j = 0; j < 4; ++j) {
      v16bf bfr = load_row32(&Bs[cur][(wn + j * 16 + l16) * kLdsStride], half);
      acc[0][j] = wmma_bf16(a0, bfr, acc[0][j]);
      acc[1][j] = wmma_bf16(a1, bfr, acc[1][j]);
    }
  }

  // epilogue: D layout => element (acc[mi][j][e], lane) = row e+8*half, col l16
#pragma unroll
  for (int j = 0; j < 4; ++j) {
    const int ncol = nbase + wn + j * 16;  // uniform per j; 16-aligned regions
    const int n = ncol + l16;
    const float bias = b_qkv[n];
    if (ncol < kD) {  // ---- Q + LoRA dq ----
      float4 lb = *(const float4*)&lora_B[n * 4];
      const int h = n >> 6, hd = n & 63;
#pragma unroll
      for (int mi = 0; mi < 2; ++mi)
#pragma unroll
        for (int e = 0; e < 8; ++e) {
          int m = mbase + wm + mi * 16 + e + 8 * half;
          float4 ta = *(const float4*)&tA[m * 8];
          float val = acc[mi][j][e] + bias +
                      kScaling * (ta.x * lb.x + ta.y * lb.y + ta.z * lb.z + ta.w * lb.w);
          int b = m >> 10, s = m & 1023;
          qbf[((b * kH + h) * kS + s) * kHD + hd] = f2bf(val);
        }
    } else if (ncol < 2 * kD) {  // ---- K ----
      const int c = n - kD, h = c >> 6, hd = c & 63;
#pragma unroll
      for (int mi = 0; mi < 2; ++mi)
#pragma unroll
        for (int e = 0; e < 8; ++e) {
          int m = mbase + wm + mi * 16 + e + 8 * half;
          float val = acc[mi][j][e] + bias;
          int b = m >> 10, s = m & 1023;
          int idx = ((b * kH + h) * kS + s) * kHD + hd;
          kout[idx] = val;           // present[0] fp32
          kbf[idx] = f2bf(val);      // bf16 for attention
        }
    } else {  // ---- V + LoRA dv ----
      const int c = n - 2 * kD, h = c >> 6, hd = c & 63;
      float4 lb = *(const float4*)&lora_B[(kD + c) * 4];
#pragma unroll
      for (int mi = 0; mi < 2; ++mi)
#pragma unroll
        for (int e = 0; e < 8; ++e) {
          int m = mbase + wm + mi * 16 + e + 8 * half;
          float4 ta = *(const float4*)&tA[m * 8 + 4];
          float val = acc[mi][j][e] + bias +
                      kScaling * (ta.x * lb.x + ta.y * lb.y + ta.z * lb.z + ta.w * lb.w);
          int b = m >> 10, s = m & 1023;
          vout[((b * kH + h) * kS + s) * kHD + hd] = val;        // present[1]
          vtbf[((b * kH + h) * kHD + hd) * kS + s] = f2bf(val);  // transposed bf16
        }
    }
  }
}

// ---------------- causal flash attention (one wave = 16 q rows) ----------------
__global__ __launch_bounds__(256) void attn_kernel(
    const __bf16* __restrict__ qbf, const __bf16* __restrict__ kbf,
    const __bf16* __restrict__ vtbf, __bf16* __restrict__ attnbf) {
  __shared__ __bf16 Psh[8][16 * kLdsStride];  // per-wave P transpose staging
  const int tid = threadIdx.x;
  const int lane = tid & 31, w = tid >> 5;
  const int half = lane >> 4, l16 = lane & 15;
  const int qblk = blockIdx.x & 7;   // S/128 = 8 q blocks
  const int bh = blockIdx.x >> 3;    // (b,h) pair 0..127
  const int b = bh >> 4, h = bh & 15;
  const int qbase = qblk * 128 + w * 16;

  // Q fragments (A layout): lane holds row qbase+l16
  v16bf qf0, qf1;
  load_row64(qbf + (bh * kS + qbase + l16) * kHD, half, qf0, qf1);

  v8f o[4] = {};
  float rm[8], rs[8];
#pragma unroll
  for (int e = 0; e < 8; ++e) { rm[e] = -1e30f; rs[e] = 0.0f; }

  const int nkt = ((qbase + 15) >> 5) + 1;  // 32-key blocks (causal bound)
  for (int kt = 0; kt < nkt; ++kt) {
    // ---- scores S(16x32) = Q(16x64) @ K^T, two 16-wide N tiles ----
    v8f sc[2];
#pragma unroll
    for (int nt = 0; nt < 2; ++nt) {
      int key = kt * 32 + nt * 16 + l16;
      v16bf kf0, kf1;
      load_row64(kbf + (bh * kS + key) * kHD, half, kf0, kf1);
      v8f z = {};
      z = wmma_bf16(qf0, kf0, z);
      z = wmma_bf16(qf1, kf1, z);
      sc[nt] = z;
    }
    // ---- causal mask + online softmax (row = e + 8*half, col = l16) ----
#pragma unroll
    for (int e = 0; e < 8; ++e) {
      int mrow = qbase + e + 8 * half;
      int key0 = kt * 32 + l16;
      float v0 = (key0 <= mrow) ? sc[0][e] * 0.125f : -1e30f;
      float v1 = (key0 + 16 <= mrow) ? sc[1][e] * 0.125f : -1e30f;
      float mx = fmaxf(v0, v1);
#pragma unroll
      for (int o2 = 1; o2 < 16; o2 <<= 1) mx = fmaxf(mx, __shfl_xor(mx, o2, 16));
      float nm = fmaxf(rm[e], mx);
      float corr = __expf(rm[e] - nm);
      float p0 = __expf(v0 - nm);
      float p1 = __expf(v1 - nm);
      float sum = p0 + p1;
#pragma unroll
      for (int o2 = 1; o2 < 16; o2 <<= 1) sum += __shfl_xor(sum, o2, 16);
      rs[e] = rs[e] * corr + sum;
      rm[e] = nm;
#pragma unroll
      for (int j = 0; j < 4; ++j) o[j][e] *= corr;
      int lrow = e + 8 * half;
      Psh[w][lrow * kLdsStride + l16] = f2bf(p0);
      Psh[w][lrow * kLdsStride + 16 + l16] = f2bf(p1);
    }
    // ---- P (D layout) -> A layout via LDS, then O += P @ V ----
    v16bf pf = load_row32(&Psh[w][l16 * kLdsStride], half);
#pragma unroll
    for (int j = 0; j < 4; ++j) {
      const __bf16* vr = vtbf + (bh * kHD + j * 16 + l16) * kS + kt * 32;
      o[j] = wmma_bf16(pf, load_row32(vr, half), o[j]);
    }
  }

  // ---- normalize + store attn in (B,S,H*HD) bf16 for the proj GEMM ----
#pragma unroll
  for (int e = 0; e < 8; ++e) {
    float inv = 1.0f / rs[e];
    int srow = qbase + e + 8 * half;
    int m = b * kS + srow;
#pragma unroll
    for (int j = 0; j < 4; ++j) {
      int col = h * kHD + j * 16 + l16;
      attnbf[m * kD + col] = f2bf(o[j][e] * inv);
    }
  }
}

// ---------------- output projection GEMM + bias ----------------
__global__ __launch_bounds__(256) void proj_gemm_kernel(
    const __bf16* __restrict__ abf, const __bf16* __restrict__ wT,
    const float* __restrict__ b_proj, float* __restrict__ out) {
  __shared__ __bf16 As[2][128 * kLdsStride];
  __shared__ __bf16 Bs[2][128 * kLdsStride];
  const int tid = threadIdx.x;
  const int lane = tid & 31, w = tid >> 5;
  const int half = lane >> 4, l16 = lane & 15;
  const int nblkN = kD / 128;
  const int mbase = (blockIdx.x / nblkN) * 128;
  const int nbase = (blockIdx.x % nblkN) * 128;
  const int wm = (w & 3) * 32;
  const int wn = (w >> 2) * 64;
  v8f acc[2][4] = {};

  constexpr int NK = kD / 32;
  stage_tiles_async(As[0], Bs[0], abf, wT, mbase, nbase, 0, tid);
  for (int kt = 0; kt < NK; ++kt) {
    const int cur = kt & 1;
    wait_async0();
    __syncthreads();
    if (kt + 1 < NK)
      stage_tiles_async(As[cur ^ 1], Bs[cur ^ 1], abf, wT, mbase, nbase,
                        (kt + 1) * 32, tid);
    v16bf a0 = load_row32(&As[cur][(wm + l16) * kLdsStride], half);
    v16bf a1 = load_row32(&As[cur][(wm + 16 + l16) * kLdsStride], half);
#pragma unroll
    for (int j = 0; j < 4; ++j) {
      v16bf bfr = load_row32(&Bs[cur][(wn + j * 16 + l16) * kLdsStride], half);
      acc[0][j] = wmma_bf16(a0, bfr, acc[0][j]);
      acc[1][j] = wmma_bf16(a1, bfr, acc[1][j]);
    }
  }
#pragma unroll
  for (int j = 0; j < 4; ++j) {
    int n = nbase + wn + j * 16 + l16;
    float bias = b_proj[n];
#pragma unroll
    for (int mi = 0; mi < 2; ++mi)
#pragma unroll
      for (int e = 0; e < 8; ++e) {
        int m = mbase + wm + mi * 16 + e + 8 * half;
        out[m * kD + n] = acc[mi][j][e] + bias;
      }
  }
}

// ---------------- host launch ----------------
extern "C" void kernel_launch(void* const* d_in, const int* in_sizes, int n_in,
                              void* d_out, int out_size, void* d_ws,
                              size_t ws_size, hipStream_t stream) {
  (void)in_sizes; (void)n_in; (void)out_size; (void)ws_size;
  const float* x      = (const float*)d_in[0];
  const float* W_qkv  = (const float*)d_in[1];
  const float* b_qkv  = (const float*)d_in[2];
  const float* lora_A = (const float*)d_in[3];
  const float* lora_B = (const float*)d_in[4];
  const float* W_proj = (const float*)d_in[5];
  const float* b_proj = (const float*)d_in[6];

  float* out  = (float*)d_out;                       // (B,S,D)
  float* kout = out + (size_t)kM * kD;               // present[0] (B,H,S,HD)
  float* vout = kout + (size_t)kM * kD;              // present[1]

  char* ws = (char*)d_ws;
  size_t off = 0;
  __bf16* xbf    = (__bf16*)(ws + off); off += (size_t)kM * kD * 2;
  __bf16* wqkvT  = (__bf16*)(ws + off); off += (size_t)kNqkv * kD * 2;
  __bf16* wprojT = (__bf16*)(ws + off); off += (size_t)kD * kD * 2;
  __bf16* qbf    = (__bf16*)(ws + off); off += (size_t)kM * kD * 2;
  __bf16* kbf    = (__bf16*)(ws + off); off += (size_t)kM * kD * 2;
  __bf16* vtbf   = (__bf16*)(ws + off); off += (size_t)kM * kD * 2;
  __bf16* attnbf = (__bf16*)(ws + off); off += (size_t)kM * kD * 2;
  float*  tA     = (float*)(ws + off);

  cvt_kernel<<<(kM * kD + 255) / 256, 256, 0, stream>>>(x, xbf, kM * kD);
  cvtT_kernel<<<(kNqkv * kD + 255) / 256, 256, 0, stream>>>(W_qkv, wqkvT, kD, kNqkv);
  cvtT_kernel<<<(kD * kD + 255) / 256, 256, 0, stream>>>(W_proj, wprojT, kD, kD);
  lora_ta_kernel<<<kM, 256, 0, stream>>>(x, lora_A, tA);
  qkv_gemm_kernel<<<(kM / 128) * (kNqkv / 128), 256, 0, stream>>>(
      xbf, wqkvT, b_qkv, lora_B, tA, qbf, kbf, vtbf, kout, vout);
  attn_kernel<<<kB * kH * (kS / 128), 256, 0, stream>>>(qbf, kbf, vtbf, attnbf);
  proj_gemm_kernel<<<(kM / 128) * (kD / 128), 256, 0, stream>>>(
      attnbf, wprojT, b_proj, out);
}
